// W2V_SM_36885179138310
// MI455X (gfx1250) — matrix operand order repaired and checked by reading, not compiled, not run on hardware
//
#include <hip/hip_runtime.h>
#include <hip/hip_bf16.h>

// Word2Vec full-softmax NLL, fused streaming-softmax GEMM for gfx1250.
//
// logits = (f16)emb_in[context] @ (f16)emb_out_w^T, accumulate exp() online,
// nll = log(sum exp) - logit_label (|logit| < 0.03 so max=0 is safe),
// label logit computed separately in f32.
//
// Main kernel: block = 8 waves; each wave owns TWO 16-row M-tiles (32 rows,
// block = 256 rows) so every B fragment read from LDS feeds two WMMAs
// (halves LDS bytes per FLOP). Weight tiles (16 cols x 256 K f16 = 8KB)
// stream into LDS via global_load_async_to_lds_b128 (ASYNCcnt), 4-buffer
// pipeline with prefetch distance 2, one barrier per tile. All 8 B fragments
// are preloaded into distinct registers (16 ds_load_b128 batched, partial
// s_wait_dscnt) before the 16-WMMA burst. Two accumulators (one per M-tile)
// chain across K so exp() reads WMMA outputs directly (no pairing movs).
// Vocab split 4 ways over blockIdx.y; partials combined in a deterministic
// single-block finalize.

#define VOCAB 32000
#define EMB 256
#define NROWS 16384           // B*W = 2048*8
#define VTILES (VOCAB / 16)   // 2000
#define VPARTS 4
#define TPP (VTILES / VPARTS) // 500 tiles per part
#define ROWS_PER_BLOCK 256    // 8 waves x 2 M-tiles x 16 rows
#define TILE_BYTES (16 * EMB * 2) // 8192

typedef __attribute__((ext_vector_type(16))) _Float16 v16h;
typedef __attribute__((ext_vector_type(8)))  _Float16 v8h;
typedef __attribute__((ext_vector_type(4)))  _Float16 v4h;
typedef __attribute__((ext_vector_type(8)))  float    v8f;

// ---------------- f32 -> f16 weight convert (once, 16MB stays in L2) -------
__global__ __launch_bounds__(256) void w2v_convert(
    const float* __restrict__ w, _Float16* __restrict__ wh)
{
  int t = blockIdx.x * 256 + threadIdx.x;         // VOCAB*EMB/4 tasks
  const float4 f = *(const float4*)(w + (size_t)t * 4);
  v4h h; h[0] = (_Float16)f.x; h[1] = (_Float16)f.y;
         h[2] = (_Float16)f.z; h[3] = (_Float16)f.w;
  *(v4h*)(wh + (size_t)t * 4) = h;
}

// ---------------- gather context embeddings -> f16 X [NROWS, EMB] ----------
__global__ __launch_bounds__(256) void w2v_gather(
    const float* __restrict__ emb_in, const int* __restrict__ context,
    _Float16* __restrict__ Xh)
{
  int t = blockIdx.x * 256 + threadIdx.x;         // NROWS*64 tasks
  int n = t >> 6;
  int k = (t & 63) * 4;
  int ctx = context[n];
  const float4 f = *(const float4*)(emb_in + (size_t)ctx * EMB + k);
  v4h h; h[0] = (_Float16)f.x; h[1] = (_Float16)f.y;
         h[2] = (_Float16)f.z; h[3] = (_Float16)f.w;
  *(v4h*)(Xh + (size_t)n * EMB + k) = h;
}

// ---------------- per-row label logit in f32 (16384 dots of length 256) ----
__global__ __launch_bounds__(256) void w2v_label(
    const float* __restrict__ emb_in, const int* __restrict__ context,
    const int* __restrict__ center, const float* __restrict__ emb_out_w,
    float* __restrict__ lbl)
{
  const int lane = threadIdx.x & 31;
  const int wave = threadIdx.x >> 5;
  const int n = blockIdx.x * 8 + wave;            // < NROWS
  const int ctx = context[n];
  const int lab = center[n >> 3];                 // W == 8
  const float* xr = emb_in + (size_t)ctx * EMB;
  const float* wr = emb_out_w + (size_t)lab * EMB;
  float acc = 0.f;
#pragma unroll
  for (int i = 0; i < 8; ++i) {
    int k = lane + i * 32;
    acc = fmaf(xr[k], wr[k], acc);
  }
  for (int off = 16; off; off >>= 1) acc += __shfl_xor(acc, off, 32);
  if (lane == 0) lbl[n] = acc;
}

// ---------------- main fused GEMM + streaming sum-exp ----------------------
__global__ __launch_bounds__(256) void w2v_main(
    const _Float16* __restrict__ Xh, const _Float16* __restrict__ Wh,
    float* __restrict__ Spart)
{
  __shared__ _Float16 sB[4][16 * EMB];            // 4 x 8KB tile buffers

  const int tid  = threadIdx.x;
  const int lane = tid & 31;
  const int wave = tid >> 5;
  const int lm   = lane & 15;
  const int kh   = lane >> 4;

  const int mblock = blockIdx.x;                  // 0..63
  const int part   = blockIdx.y;                  // 0..3
  const int t0     = part * TPP;

  // ---- A fragments: 2 M-tiles x 16 rows x K=256, in registers for the
  // whole kernel. A layout matches B: lane(l<16) row=l holds
  // K 0..15 contiguous, lanes 16..31 hold K 16..31 per K=32 step.
  v16h afrag[2][8];
#pragma unroll
  for (int mt = 0; mt < 2; ++mt) {
    const int arow = mblock * ROWS_PER_BLOCK + wave * 32 + mt * 16 + lm;
    const _Float16* aptr = Xh + (size_t)arow * EMB;
#pragma unroll
    for (int kc = 0; kc < 8; ++kc) {
      v8h lo = *(const v8h*)(aptr + kc * 32 + kh * 16);
      v8h hi = *(const v8h*)(aptr + kc * 32 + kh * 16 + 8);
      afrag[mt][kc] = __builtin_shufflevector(lo, hi,
          0, 1, 2, 3, 4, 5, 6, 7, 8, 9, 10, 11, 12, 13, 14, 15);
    }
  }

  // ---- async-copy setup: thread copies 32B of each 8KB tile (2 x B128)
  const int crow = tid >> 4;                      // tile row 0..15
  const int cseg = tid & 15;                      // 16-half segment
  const unsigned chunkOff = (unsigned)((crow * EMB + cseg * 16) * 2);
  const unsigned ldsBase = (unsigned)(uintptr_t)(&sB[0][0]) + chunkOff;

  auto issue_tile = [&](int ti, int buf) {
    unsigned g0 = (unsigned)(t0 + ti) * (unsigned)TILE_BYTES + chunkOff;
    unsigned l0 = ldsBase + (unsigned)buf * (unsigned)TILE_BYTES;
    asm volatile("global_load_async_to_lds_b128 %0, %1, %2"
                 :: "v"(l0), "v"(g0), "s"(Wh) : "memory");
    asm volatile("global_load_async_to_lds_b128 %0, %1, %2"
                 :: "v"(l0 + 16u), "v"(g0 + 16u), "s"(Wh) : "memory");
  };

  float sacc[2][8];
#pragma unroll
  for (int mt = 0; mt < 2; ++mt)
#pragma unroll
    for (int g = 0; g < 8; ++g) sacc[mt][g] = 0.f;

  issue_tile(0, 0);
  issue_tile(1, 1);

  for (int i = 0; i < TPP; ++i) {
    if (i + 2 < TPP) {
      issue_tile(i + 2, (i + 2) & 3);
      // outstanding per wave: tiles i,i+1,i+2 (2 instr each, in-order)
      // -> wait <=4 guarantees tile i is in LDS.
      asm volatile("s_wait_asynccnt 4" ::: "memory");
    } else {
      asm volatile("s_wait_asynccnt 0" ::: "memory");
    }
    __syncthreads();  // publish tile i; protects buffer (i+2)&3 reuse
                      // (last read at compute(i-2), >=1 barrier ago)

    // B layout: lane(l<16) col=l holds K 0..15 contiguous, lanes 16..31
    // hold K 16..31 -> one contiguous 32B (v16h) per lane per K=32 step.
    const _Float16* bp = &sB[i & 3][lm * EMB + kh * 16];

    // Preload all 8 B fragments (16 ds_load_b128, batched) ...
    v16h b[8];
#pragma unroll
    for (int kc = 0; kc < 8; ++kc)
      b[kc] = *(const v16h*)(bp + kc * 32);

    // ... then a 16-WMMA burst: two interleaved accumulator chains.
    v8f c0 = {0.f,0.f,0.f,0.f,0.f,0.f,0.f,0.f};   // M-tile 0
    v8f c1 = {0.f,0.f,0.f,0.f,0.f,0.f,0.f,0.f};   // M-tile 1
#pragma unroll
    for (int kc = 0; kc < 8; ++kc) {
      c0 = __builtin_amdgcn_wmma_f32_16x16x32_f16(false, afrag[0][kc], false, b[kc],
                                                  (short)0, c0, false, false);
      c1 = __builtin_amdgcn_wmma_f32_16x16x32_f16(false, afrag[1][kc], false, b[kc],
                                                  (short)0, c1, false, false);
    }

#pragma unroll
    for (int g = 0; g < 8; ++g) {
      sacc[0][g] += __expf(c0[g]);
      sacc[1][g] += __expf(c1[g]);
    }
  }

  // ---- per-row reduction: C vgpr g / lane l -> row g + 8*(l>>4), col l&15.
  // Sum across the 16 lanes of each half-wave (xor 1,2,4,8 stays in-half).
#pragma unroll
  for (int mt = 0; mt < 2; ++mt) {
#pragma unroll
    for (int g = 0; g < 8; ++g) {
      float v = sacc[mt][g];
      v += __shfl_xor(v, 1, 32);
      v += __shfl_xor(v, 2, 32);
      v += __shfl_xor(v, 4, 32);
      v += __shfl_xor(v, 8, 32);
      sacc[mt][g] = v;
    }
  }
  if (lm == 0) {
#pragma unroll
    for (int mt = 0; mt < 2; ++mt)
#pragma unroll
      for (int g = 0; g < 8; ++g) {
        int row = mblock * ROWS_PER_BLOCK + wave * 32 + mt * 16 + g + 8 * kh;
        Spart[(size_t)part * NROWS + row] = sacc[mt][g];
      }
  }
}

// ---------------- combine vocab parts, nll, mean (single block, determin.) -
__global__ __launch_bounds__(1024) void w2v_finalize(
    const float* __restrict__ Spart, const float* __restrict__ lbl,
    float* __restrict__ out)
{
  float acc = 0.f;
  for (int row = threadIdx.x; row < NROWS; row += 1024) {
    float S = Spart[row] + Spart[NROWS + row] +
              Spart[2 * NROWS + row] + Spart[3 * NROWS + row];
    acc += logf(S) - lbl[row];       // nll with implicit max=0 (|logit|<0.03)
  }
  for (int off = 16; off; off >>= 1) acc += __shfl_xor(acc, off, 32);
  __shared__ float red[32];
  if ((threadIdx.x & 31) == 0) red[threadIdx.x >> 5] = acc;
  __syncthreads();
  if (threadIdx.x < 32) {
    float v = red[threadIdx.x];
    for (int off = 16; off; off >>= 1) v += __shfl_xor(v, off, 32);
    if (threadIdx.x == 0) out[0] = v * (1.0f / NROWS);
  }
}

extern "C" void kernel_launch(void* const* d_in, const int* in_sizes, int n_in,
                              void* d_out, int out_size, void* d_ws, size_t ws_size,
                              hipStream_t stream) {
  const int*   center    = (const int*)d_in[0];
  const int*   context   = (const int*)d_in[1];
  const float* emb_in    = (const float*)d_in[2];
  const float* emb_out_w = (const float*)d_in[3];
  float* out = (float*)d_out;

  // workspace layout (all 256B aligned): Wh 16,384,000 | Xh 8,388,608 |
  // lbl 65,536 | Spart 262,144  (~25.1 MB total)
  char* ws = (char*)d_ws;
  _Float16* Wh  = (_Float16*)ws;
  _Float16* Xh  = (_Float16*)(ws + (size_t)VOCAB * EMB * 2);
  float*    lbl = (float*)(ws + (size_t)(VOCAB + NROWS) * EMB * 2);
  float*    Sp  = (float*)((char*)lbl + (size_t)NROWS * 4);

  w2v_convert <<<VOCAB * EMB / 4 / 256, 256, 0, stream>>>(emb_out_w, Wh);
  w2v_gather  <<<NROWS * EMB / 4 / 256, 256, 0, stream>>>(emb_in, context, Xh);
  w2v_label   <<<NROWS / 8,             256, 0, stream>>>(emb_in, context, center,
                                                          emb_out_w, lbl);
  w2v_main    <<<dim3(NROWS / ROWS_PER_BLOCK, VPARTS), 256, 0, stream>>>(Xh, Wh, Sp);
  w2v_finalize<<<1, 1024, 0, stream>>>(Sp, lbl, out);
}
